// Model_baseline_44315472560258
// MI455X (gfx1250) — matrix operand order repaired and checked by reading, not compiled, not run on hardware
//
#include <hip/hip_runtime.h>
#include <hip/hip_bf16.h>
#include <math.h>

// ---------------------------------------------------------------------------
// GCN (3x GCNConv, sym-norm w/ self loops) + 2-layer MLP head + log_softmax.
// GEMMs via v_wmma_f32_16x16x32_bf16 (f32 accumulate); edge aggregation via
// coalesced gather + global_atomic_add_f32 scatter (memory-bound part).
// Weights pre-packed so B fragments are two b128 loads per K-step; A
// fragments are two ds_load_b128 per K-step. Epilogue flags are template
// parameters (no per-row exec-mask branches; N is a multiple of 16).
// ---------------------------------------------------------------------------

typedef __attribute__((ext_vector_type(16))) __bf16   v16bf;
typedef __attribute__((ext_vector_type(8)))  float    v8f;
typedef __attribute__((ext_vector_type(4)))  unsigned uv4;
typedef __attribute__((ext_vector_type(4)))  float    fv4;
typedef __attribute__((ext_vector_type(2)))  float    fv2;

union FragB16 {            // 8 VGPRs = 16 bf16 values per lane
    unsigned u[8];
    uv4      q[2];
    v16bf    v;
};

__device__ __forceinline__ unsigned short f2bf(float f) {
    unsigned u = __float_as_uint(f);
    unsigned r = u + 0x7FFFu + ((u >> 16) & 1u);   // round-to-nearest-even
    return (unsigned short)(r >> 16);
}
__device__ __forceinline__ unsigned packbf(float lo, float hi) {
    return (unsigned)f2bf(lo) | ((unsigned)f2bf(hi) << 16);
}

// ---------------- small utility kernels ------------------------------------

__global__ void k_fill(float* p, float v, int n) {
    int i = blockIdx.x * blockDim.x + threadIdx.x;
    if (i < n) p[i] = v;
}

// out[i][j] = bias[j]   (pre-scatter init so aggregation lands on bias)
__global__ void k_bias_init(float* out, const float* b, int nrows) {
    int i = blockIdx.x * blockDim.x + threadIdx.x;
    if (i < nrows * 128) out[i] = b[i & 127];
}

// Pack W[128][ncols] f32 -> B-fragment-friendly layout:
//   pair-row r (0..63) packs bf16(W[2r][n]), bf16(W[2r+1][n]) (K even in lo16)
//   stored at uint index ((r>>2)*ncolsP + n)*4 + (r&3)
// so rows r0..r0+3 of one column are a contiguous uint4 (b128 load).
__global__ void k_pack_w(const float* W, unsigned* Wp, int ncols, int ncolsP) {
    int o = blockIdx.x * blockDim.x + threadIdx.x;
    if (o >= 64 * ncolsP) return;
    int s   = o & 3;
    int t   = o >> 2;
    int n   = t % ncolsP;
    int g   = t / ncolsP;
    int r   = g * 4 + s;                 // pair-row index
    float a = (n < ncols) ? W[(2 * r)     * ncols + n] : 0.0f;
    float b = (n < ncols) ? W[(2 * r + 1) * ncols + n] : 0.0f;
    Wp[o] = packbf(a, b);
}

// deg counting over target nodes (self-loop +1 folded into finalize)
__global__ void k_count_deg(const int* col, float* deg, int E) {
    int e = blockIdx.x * blockDim.x + threadIdx.x;
    if (e < E) atomicAdd(&deg[col[e]], 1.0f);
}

__global__ void k_dinv(float* d, int n) {
    int i = blockIdx.x * blockDim.x + threadIdx.x;
    if (i < n) d[i] = rsqrtf(d[i] + 1.0f);   // +1 for self loop; always > 0
}

// norm[e] = dinv[row]*dinv[col]; tail (e >= E) are self loops: dinv[i]^2
__global__ void k_norm(const int* row, const int* col, const float* dinv,
                       float* nrm, int E, int N) {
    int e = blockIdx.x * blockDim.x + threadIdx.x;
    if (e >= E + N) return;
    if (e < E) nrm[e] = dinv[row[e]] * dinv[col[e]];
    else { int i = e - E; nrm[e] = dinv[i] * dinv[i]; }
}

// ---------------- WMMA GEMM: Y[N,128] = act(X)[N,128] @ Wbf16[128,128] ------
// block = 256 thr (8 waves); block handles 16 rows; wave w -> 16-col tile w.
// LDS stages the 16x128 X tile as packed bf16 (ReLU fused into conversion).
// Requires nrows % 16 == 0 for the store epilogue (true here: 50000 = 3125*16).
template <int RELU_IN, int RELU_OUT, int HAS_BIAS>
__global__ void __launch_bounds__(256)
k_gemm128(const float* __restrict__ X, const unsigned* __restrict__ Wp,
          float* __restrict__ Y, const float* __restrict__ bias, int nrows) {
    __shared__ unsigned ldsX[16 * 64];          // 16 rows x 64 bf16-pairs

    const int t    = threadIdx.x;
    const int row0 = blockIdx.x * 16;

    // ---- stage X tile -> LDS (bf16 pairs), 8 floats per thread ----
    {
        int r  = t >> 4;
        int c8 = (t & 15) * 8;
        const float* src = X + (size_t)(row0 + r) * 128 + c8;
        fv4 f0 = *(const fv4*)(src);
        fv4 f1 = *(const fv4*)(src + 4);
        if (RELU_IN) {
            f0.x = fmaxf(f0.x, 0.f); f0.y = fmaxf(f0.y, 0.f);
            f0.z = fmaxf(f0.z, 0.f); f0.w = fmaxf(f0.w, 0.f);
            f1.x = fmaxf(f1.x, 0.f); f1.y = fmaxf(f1.y, 0.f);
            f1.z = fmaxf(f1.z, 0.f); f1.w = fmaxf(f1.w, 0.f);
        }
        unsigned* dst = &ldsX[r * 64 + (c8 >> 1)];
        dst[0] = packbf(f0.x, f0.y);
        dst[1] = packbf(f0.z, f0.w);
        dst[2] = packbf(f1.x, f1.y);
        dst[3] = packbf(f1.z, f1.w);
    }
    __syncthreads();

    // ---- per-wave 16x16 tile ----
    const int wave = t >> 5;
    const int lane = t & 31;
    const int m    = lane & 15;      // A row / C column-lane
    const int kh   = lane >> 4;      // lane half
    const int ncol = wave * 16 + m;  // global output column

    v8f acc = {};
    const uv4* l4 = (const uv4*)ldsX;
    const uv4* w4 = (const uv4*)Wp;

    #pragma unroll
    for (int kb = 0; kb < 128; kb += 32) {
        FragB16 A, B;
        // A fragment (16-bit A-matrix 16x32 layout): two b128 LDS loads
        int a4 = m * 16 + (kb >> 3) + kh;
        A.q[0] = l4[a4];
        A.q[1] = l4[a4 + 2];
        // B fragment: v_i packs K = kb + kh*8 + 2i (i<4), +16 for i>=4; packed layout gives
        // pair-rows r0..r0+3 / r0+8..r0+11 of column ncol as two uint4s.
        int g = ((kb >> 1) + kh * 4) >> 2;       // r0/4
        B.q[0] = w4[(g + 0) * 128 + ncol];
        B.q[1] = w4[(g + 2) * 128 + ncol];

        acc = __builtin_amdgcn_wmma_f32_16x16x32_bf16(
                false, A.v, false, B.v, (short)0, acc, false, false);
    }

    // ---- epilogue: compile-time bias/relu, straight f32 stores ----
    float bv = HAS_BIAS ? bias[ncol] : 0.0f;
    float* dst = Y + (size_t)(row0 + kh * 8) * 128 + ncol;
    #pragma unroll
    for (int j = 0; j < 8; ++j) {
        float v = acc[j] + bv;
        if (RELU_OUT) v = fmaxf(v, 0.0f);
        dst[(size_t)j * 128] = v;               // C layout: M = j + 8*half
    }
}

// ---------------- edge scatter: OUT[col] += norm * XW[row] ------------------
// one wave per edge-slot (E real edges + N self loops), 4 floats per lane
__global__ void __launch_bounds__(256)
k_scatter(const float* __restrict__ XW, const float* __restrict__ nrm,
          const int* __restrict__ row, const int* __restrict__ col,
          float* __restrict__ OUT, int E, int N) {
    int idx  = blockIdx.x * blockDim.x + threadIdx.x;
    int e    = idx >> 5;
    int lane = idx & 31;
    if (e >= E + N) return;
    int r, c;
    float w = nrm[e];
    if (e < E) { r = row[e]; c = col[e]; }
    else       { r = c = e - E; }
    int j = lane * 4;
    fv4 v = *(const fv4*)(XW + (size_t)r * 128 + j);
    float* dst = OUT + (size_t)c * 128 + j;
    atomicAdd(dst + 0, v.x * w);
    atomicAdd(dst + 1, v.y * w);
    atomicAdd(dst + 2, v.z * w);
    atomicAdd(dst + 3, v.w * w);
}

// ---------------- head2 + log_softmax: z1[N,128] @ lW2[128,10] --------------
// one wave per 16-row tile; 16-col padded WMMA, cols >= 10 masked out;
// log_softmax via width-16 shfl_xor reductions (row lives across 16 lanes).
__global__ void __launch_bounds__(32)
k_head2_lsm(const float* __restrict__ Z1, const unsigned* __restrict__ Wp,
            const float* __restrict__ b2, float* __restrict__ out0,
            float* __restrict__ out1) {
    const int lane = threadIdx.x;
    const int m    = lane & 15;
    const int kh   = lane >> 4;
    const int row0 = blockIdx.x * 16;

    const uv4* w4 = (const uv4*)Wp;
    v8f acc = {};
    #pragma unroll
    for (int kb = 0; kb < 128; kb += 32) {
        FragB16 A, B;
        const float* src = Z1 + (size_t)(row0 + m) * 128;
        #pragma unroll
        for (int i = 0; i < 4; ++i) {
            int k = kb + kh * 8 + 2 * i;
            fv2 f = *(const fv2*)(src + k);
            fv2 g = *(const fv2*)(src + k + 16);
            A.u[i]     = packbf(f.x, f.y);
            A.u[4 + i] = packbf(g.x, g.y);
        }
        int g = ((kb >> 1) + kh * 4) >> 2;
        B.q[0] = w4[(g + 0) * 16 + m];
        B.q[1] = w4[(g + 2) * 16 + m];

        acc = __builtin_amdgcn_wmma_f32_16x16x32_bf16(
                false, A.v, false, B.v, (short)0, acc, false, false);
    }

    float bv = (m < 10) ? b2[m] : 0.0f;
    #pragma unroll
    for (int j = 0; j < 8; ++j) {
        float z = (m < 10) ? (acc[j] + bv) : -INFINITY;
        float mx = z;
        #pragma unroll
        for (int s = 1; s < 16; s <<= 1)
            mx = fmaxf(mx, __shfl_xor(mx, s, 16));
        float ex = (m < 10) ? __expf(z - mx) : 0.0f;
        float se = ex;
        #pragma unroll
        for (int s = 1; s < 16; s <<= 1)
            se += __shfl_xor(se, s, 16);
        float ls = z - mx - __logf(se);
        if (m < 10) {
            int r = row0 + j + kh * 8;
            out0[r * 10 + m] = ls;
            out1[r * 10 + m] = ls;
        }
    }
}

// ---------------------------------------------------------------------------
extern "C" void kernel_launch(void* const* d_in, const int* in_sizes, int n_in,
                              void* d_out, int out_size, void* d_ws, size_t ws_size,
                              hipStream_t stream) {
    const float* x    = (const float*)d_in[0];
    const int*   ei   = (const int*)  d_in[1];     // [2][E] int32
    const float* W0   = (const float*)d_in[3];
    const float* b0   = (const float*)d_in[4];
    const float* W1   = (const float*)d_in[5];
    const float* b1   = (const float*)d_in[6];
    const float* W2   = (const float*)d_in[7];
    const float* b2   = (const float*)d_in[8];
    const float* lW1  = (const float*)d_in[9];
    const float* lb1  = (const float*)d_in[10];
    const float* lW2  = (const float*)d_in[11];
    const float* lb2  = (const float*)d_in[12];

    const int N = in_sizes[0] / 128;               // 50000 (multiple of 16)
    const int E = in_sizes[1] / 2;                 // 800000
    const int* erow = ei;
    const int* ecol = ei + E;

    // ---- workspace layout (~55 MB) ----
    float* ws   = (float*)d_ws;
    float* deg  = ws;                              // N   (deg -> dinv in place)
    float* nrm  = deg + N;                         // E+N
    float* xw   = nrm + (size_t)(E + N);           // N*128 (GEMM out / z1)
    float* hbuf = xw + (size_t)N * 128;            // N*128 (aggregation)
    unsigned* wp   = (unsigned*)(hbuf + (size_t)N * 128);
    unsigned* Wp0  = wp;
    unsigned* Wp1  = wp + 8192;
    unsigned* Wp2  = wp + 16384;
    unsigned* WpL1 = wp + 24576;
    unsigned* WpL2 = wp + 32768;                   // 64*16

    float* out_lsm  = (float*)d_out;               // [N,10]
    float* out_h    = out_lsm + (size_t)N * 10;    // [N,128]
    float* out_lsm2 = out_h + (size_t)N * 128;     // [N,10]

    const int TB = 256;
    const int rowTiles = N / 16;                   // 3125

    // ---- weight packing (once per launch; tiny) ----
    k_pack_w<<<(64 * 128 + TB - 1) / TB, TB, 0, stream>>>(W0,  Wp0, 128, 128);
    k_pack_w<<<(64 * 128 + TB - 1) / TB, TB, 0, stream>>>(W1,  Wp1, 128, 128);
    k_pack_w<<<(64 * 128 + TB - 1) / TB, TB, 0, stream>>>(W2,  Wp2, 128, 128);
    k_pack_w<<<(64 * 128 + TB - 1) / TB, TB, 0, stream>>>(lW1, WpL1, 128, 128);
    k_pack_w<<<(64 * 16  + TB - 1) / TB, TB, 0, stream>>>(lW2, WpL2, 10, 16);

    // ---- GCN normalization ----
    k_fill<<<(N + TB - 1) / TB, TB, 0, stream>>>(deg, 0.0f, N);
    k_count_deg<<<(E + TB - 1) / TB, TB, 0, stream>>>(ecol, deg, E);
    k_dinv<<<(N + TB - 1) / TB, TB, 0, stream>>>(deg, N);
    k_norm<<<(E + N + TB - 1) / TB, TB, 0, stream>>>(erow, ecol, deg, nrm, E, N);

    const int scatterBlocks = ((E + N) * 32 + TB - 1) / TB;
    const int biasBlocks    = (N * 128 + TB - 1) / TB;

    // ---- layer 0 ----
    k_gemm128<0,0,0><<<rowTiles, TB, 0, stream>>>(x, Wp0, xw, nullptr, N);
    k_bias_init<<<biasBlocks, TB, 0, stream>>>(hbuf, b0, N);
    k_scatter<<<scatterBlocks, TB, 0, stream>>>(xw, nrm, erow, ecol, hbuf, E, N);

    // ---- layer 1 (ReLU fused into GEMM input conversion) ----
    k_gemm128<1,0,0><<<rowTiles, TB, 0, stream>>>(hbuf, Wp1, xw, nullptr, N);
    k_bias_init<<<biasBlocks, TB, 0, stream>>>(hbuf, b1, N);
    k_scatter<<<scatterBlocks, TB, 0, stream>>>(xw, nrm, erow, ecol, hbuf, E, N);

    // ---- layer 2 -> aggregates directly into d_out h-slice (no ReLU) ----
    k_gemm128<1,0,0><<<rowTiles, TB, 0, stream>>>(hbuf, Wp2, xw, nullptr, N);
    k_bias_init<<<biasBlocks, TB, 0, stream>>>(out_h, b2, N);
    k_scatter<<<scatterBlocks, TB, 0, stream>>>(xw, nrm, erow, ecol, out_h, E, N);

    // ---- MLP head: z1 = relu(h @ lW1 + lb1) (reuse xw buffer) ----
    k_gemm128<0,1,1><<<rowTiles, TB, 0, stream>>>(out_h, WpL1, xw, lb1, N);

    // ---- head2 + log_softmax -> both output copies ----
    k_head2_lsm<<<rowTiles, 32, 0, stream>>>(xw, WpL2, lb2, out_lsm, out_lsm2);
}
